// PointTransformerSemanticSegmentation_61864708931981
// MI455X (gfx1250) — matrix operand order repaired and checked by reading, not compiled, not run on hardware
//
#include <hip/hip_runtime.h>

typedef __attribute__((ext_vector_type(16))) _Float16 v16h;
typedef __attribute__((ext_vector_type(8)))  _Float16 h8;
typedef __attribute__((ext_vector_type(8)))  float    v8f;

#define KNN_K 16
#define EPSF  1e-5f

// ---------------------------------------------------------------------------
// Operand packing for branch-free WMMA GEMM.
// pack_a: X(M,K) f32 row-major -> Xp(Mpad,Kpad) f16 row-major, zero padded.
// pack_bt: W(K,Nd) f32 row-major -> Wt(Npad,Kpad) f16 (transposed), zero padded.
// ---------------------------------------------------------------------------
__global__ void pack_a_kernel(const float* __restrict__ X, _Float16* __restrict__ Xp,
                              int M, int K, int Kpad, size_t tot)
{
    size_t t = (size_t)blockIdx.x * blockDim.x + threadIdx.x;
    if (t >= tot) return;
    int k = (int)(t % Kpad);
    size_t r = t / Kpad;
    Xp[t] = (r < (size_t)M && k < K) ? (_Float16)X[r * K + k] : (_Float16)0.0f;
}

__global__ void pack_bt_kernel(const float* __restrict__ W, _Float16* __restrict__ Wt,
                               int K, int Nd, int Kpad, size_t tot)
{
    size_t t = (size_t)blockIdx.x * blockDim.x + threadIdx.x;
    if (t >= tot) return;
    int k = (int)(t % Kpad);
    size_t c = t / Kpad;
    Wt[t] = (c < (size_t)Nd && k < K) ? (_Float16)W[(size_t)k * Nd + c] : (_Float16)0.0f;
}

// ---------------------------------------------------------------------------
// GEMM: Y(M,Nd) = X(M,K) @ W(K,Nd) + bias [+ res] [relu]
// Packed f16 operands; each wave computes a 32x16 strip (2 accumulators,
// shared B fragment) -> two v_wmma_f32_16x16x32_f16 per K-step, all loads
// contiguous b128, no divergence in the hot loop.
// ---------------------------------------------------------------------------
__global__ void gemm_wmma_kernel(const _Float16* __restrict__ Xp,
                                 const _Float16* __restrict__ Wt,
                                 const float* __restrict__ bias,
                                 const float* __restrict__ res,
                                 float* __restrict__ Y,
                                 int M, int Kpad, int Nd, int relu)
{
    const int lane  = threadIdx.x & 31;
    const int wave  = threadIdx.x >> 5;
    const int tileM = (blockIdx.x * 4 + wave) * 32;     // 32 rows per wave
    const int tileN = blockIdx.y * 16;
    if (tileM >= M) return;          // wave-uniform exit: EXEC stays all-ones
    const int hgrp = lane >> 4;      // lane half (0|1)
    const int l15  = lane & 15;
    const int col  = tileN + l15;    // < Npad always (grid covers Npad)

    const _Float16* a0p = Xp + (size_t)(tileM + l15) * Kpad + hgrp * 16;
    const _Float16* a1p = a0p + (size_t)16 * Kpad;
    const _Float16* bp  = Wt + (size_t)col * Kpad + hgrp * 16;

    v8f acc0, acc1;
    {
        float bv = (bias != nullptr && col < Nd) ? bias[col] : 0.0f;
#pragma unroll
        for (int r = 0; r < 8; ++r) { acc0[r] = bv; acc1[r] = bv; }
    }

    for (int k0 = 0; k0 < Kpad; k0 += 32) {
        v16h b = *(const v16h*)(bp + k0);               // 16 contiguous K for col
        h8 x0 = *(const h8*)(a0p + k0);                 // K sub-block 0 (8 halfs)
        h8 x1 = *(const h8*)(a0p + k0 + 8);             // K sub-block 1
        h8 y0 = *(const h8*)(a1p + k0);
        h8 y1 = *(const h8*)(a1p + k0 + 8);
        v16h a0 = __builtin_shufflevector(x0, x1, 0, 1, 2, 3, 4, 5, 6, 7,
                                          8, 9, 10, 11, 12, 13, 14, 15);
        v16h a1 = __builtin_shufflevector(y0, y1, 0, 1, 2, 3, 4, 5, 6, 7,
                                          8, 9, 10, 11, 12, 13, 14, 15);
        acc0 = __builtin_amdgcn_wmma_f32_16x16x32_f16(false, a0, false, b,
                                                      (short)0, acc0, false, false);
        acc1 = __builtin_amdgcn_wmma_f32_16x16x32_f16(false, a1, false, b,
                                                      (short)0, acc1, false, false);
    }

    const bool colOK = (col < Nd);
#pragma unroll
    for (int r = 0; r < 8; ++r) {
        int r0 = tileM + hgrp * 8 + r;
        int r1 = r0 + 16;
        if (r0 < M && colOK) {
            float v = acc0[r];
            if (res)  v += res[(size_t)r0 * Nd + col];
            if (relu) v = fmaxf(v, 0.0f);
            Y[(size_t)r0 * Nd + col] = v;
        }
        if (r1 < M && colOK) {
            float v = acc1[r];
            if (res)  v += res[(size_t)r1 * Nd + col];
            if (relu) v = fmaxf(v, 0.0f);
            Y[(size_t)r1 * Nd + col] = v;
        }
    }
}

// ---------------------------------------------------------------------------
// KNN: per query thread, scan all Nn points, keep 16 smallest squared dists.
// ---------------------------------------------------------------------------
__global__ void knn_kernel(const float* __restrict__ x, int Bp, int Nn, int C,
                           int Nq, int qstride, int* __restrict__ idx)
{
    int t = blockIdx.x * blockDim.x + threadIdx.x;
    if (t >= Bp * Nq) return;
    int b = t / Nq, q = t - b * Nq;
    const float* xb = x + (size_t)b * Nn * C;
    const float* xq = xb + (size_t)(q * qstride) * C;

    float bd[KNN_K]; int bi[KNN_K];
#pragma unroll
    for (int j = 0; j < KNN_K; ++j) { bd[j] = 3.0e38f; bi[j] = 0; }

    for (int n = 0; n < Nn; ++n) {
        const float* xn = xb + (size_t)n * C;
        if (n + 1 < Nn) __builtin_prefetch(xn + C, 0, 1);
        float d2 = 0.0f;
        for (int c = 0; c < C; ++c) {
            float df = xq[c] - xn[c];
            d2 = fmaf(df, df, d2);
        }
        if (d2 < bd[KNN_K - 1]) {
            int j = KNN_K - 1;
            while (j > 0 && d2 < bd[j - 1]) {
                bd[j] = bd[j - 1]; bi[j] = bi[j - 1]; --j;
            }
            bd[j] = d2; bi[j] = n;
        }
    }
    int* o = idx + (size_t)t * KNN_K;
#pragma unroll
    for (int j = 0; j < KNN_K; ++j) o[j] = bi[j];
}

// ---------------------------------------------------------------------------
__global__ void gather_kernel(const float* __restrict__ x, const int* __restrict__ idx,
                              float* __restrict__ y, int Bp, int Nn, int Nq, int C)
{
    size_t tot = (size_t)Bp * Nq * KNN_K * C;
    size_t t = (size_t)blockIdx.x * blockDim.x + threadIdx.x;
    if (t >= tot) return;
    int c = (int)(t % C); size_t r = t / C;
    int k = (int)(r % KNN_K); size_t bm = r / KNN_K;
    int b = (int)(bm / Nq);
    int j = idx[bm * KNN_K + k];
    y[t] = x[((size_t)b * Nn + j) * C + c];
}

// Column mean & rsqrt(var+eps) over M rows (BatchNorm1d training stats).
__global__ void colstats_kernel(const float* __restrict__ h, int M, int C,
                                float* __restrict__ mu, float* __restrict__ rs)
{
    __shared__ float s1[256], s2[256];
    int c = blockIdx.x;
    float a = 0.0f, b = 0.0f;
    for (int m = threadIdx.x; m < M; m += blockDim.x) {
        float v = h[(size_t)m * C + c];
        a += v; b += v * v;
    }
    s1[threadIdx.x] = a; s2[threadIdx.x] = b;
    __syncthreads();
    for (int s = blockDim.x >> 1; s > 0; s >>= 1) {
        if ((int)threadIdx.x < s) {
            s1[threadIdx.x] += s1[threadIdx.x + s];
            s2[threadIdx.x] += s2[threadIdx.x + s];
        }
        __syncthreads();
    }
    if (threadIdx.x == 0) {
        float m1  = s1[0] / (float)M;
        float var = s2[0] / (float)M - m1 * m1;
        mu[c] = m1;
        rs[c] = rsqrtf(var + EPSF);
    }
}

__global__ void bnrelu_kernel(const float* __restrict__ h,
                              const float* __restrict__ mu, const float* __restrict__ rs,
                              const float* __restrict__ gamma, const float* __restrict__ beta,
                              float* __restrict__ y, int M, int C)
{
    size_t tot = (size_t)M * C;
    size_t t = (size_t)blockIdx.x * blockDim.x + threadIdx.x;
    if (t >= tot) return;
    int c = (int)(t % C);
    float v = (h[t] - mu[c]) * rs[c] * gamma[c] + beta[c];
    y[t] = fmaxf(v, 0.0f);
}

__global__ void maxk_kernel(const float* __restrict__ h, float* __restrict__ y,
                            int Mq, int C)
{
    size_t tot = (size_t)Mq * C;
    size_t t = (size_t)blockIdx.x * blockDim.x + threadIdx.x;
    if (t >= tot) return;
    int c = (int)(t % C); size_t bm = t / C;
    float m = -3.0e38f;
#pragma unroll
    for (int k = 0; k < KNN_K; ++k)
        m = fmaxf(m, h[(bm * KNN_K + k) * C + c]);
    y[t] = m;
}

__global__ void possub_kernel(const float* __restrict__ pos, float* __restrict__ pout,
                              int Bp, int Nn, int n2, int stride)
{
    size_t tot = (size_t)Bp * n2 * 3;
    size_t t = (size_t)blockIdx.x * blockDim.x + threadIdx.x;
    if (t >= tot) return;
    int j = (int)(t % 3); size_t r = t / 3;
    int i = (int)(r % n2); int b = (int)(r / n2);
    pout[t] = pos[((size_t)b * Nn + (size_t)i * stride) * 3 + j];
}

__global__ void relpos_kernel(const float* __restrict__ pos, const int* __restrict__ idx,
                              float* __restrict__ rel, int Bp, int Np)
{
    size_t tot = (size_t)Bp * Np * KNN_K * 3;
    size_t t = (size_t)blockIdx.x * blockDim.x + threadIdx.x;
    if (t >= tot) return;
    int j = (int)(t % 3); size_t r = t / 3;
    int k = (int)(r % KNN_K); size_t bn = r / KNN_K;
    int b = (int)(bn / Np);
    int jj = idx[bn * KNN_K + k];
    rel[t] = pos[bn * 3 + j] - pos[((size_t)b * Np + jj) * 3 + j];
}

// Per-channel softmax over the K neighbors: out = sum_k softmax(q-k+pe)*(v+pe)
__global__ void attn_kernel(const float* __restrict__ q, const float* __restrict__ kf,
                            const float* __restrict__ vv, const float* __restrict__ pe,
                            const int* __restrict__ idx, float* __restrict__ out,
                            int Bp, int Np, int C)
{
    size_t tot = (size_t)Bp * Np * C;
    size_t t = (size_t)blockIdx.x * blockDim.x + threadIdx.x;
    if (t >= tot) return;
    int c = (int)(t % C); size_t bn = t / C;
    int b = (int)(bn / Np);
    const int* id = idx + bn * KNN_K;
    float qv = q[bn * C + c];

    float lg[KNN_K];
    float mx = -3.0e38f;
#pragma unroll
    for (int k = 0; k < KNN_K; ++k) {
        int j = id[k];
        float l = qv - kf[((size_t)b * Np + j) * C + c] + pe[(bn * KNN_K + k) * C + c];
        lg[k] = l;
        mx = fmaxf(mx, l);
    }
    float den = 0.0f, num = 0.0f;
#pragma unroll
    for (int k = 0; k < KNN_K; ++k) {
        int j = id[k];
        float e = __expf(lg[k] - mx);
        den += e;
        num += e * (vv[((size_t)b * Np + j) * C + c] + pe[(bn * KNN_K + k) * C + c]);
    }
    out[t] = num / den;
}

// 4x linear upsample along point axis (half-pixel centers, edge replicate)
__global__ void upsample4_kernel(const float* __restrict__ x, float* __restrict__ y,
                                 int Bp, int Min, int C)
{
    int Mout = Min * 4;
    size_t tot = (size_t)Bp * Mout * C;
    size_t t = (size_t)blockIdx.x * blockDim.x + threadIdx.x;
    if (t >= tot) return;
    int c = (int)(t % C); size_t r = t / C;
    int i = (int)(r % Mout); int b = (int)(r / Mout);
    float f  = (i + 0.5f) * 0.25f - 0.5f;
    float fl = floorf(f);
    int   i0 = (int)fl;
    float w  = f - fl;
    int ia = i0 < 0 ? 0 : (i0 > Min - 1 ? Min - 1 : i0);
    int ic = i0 + 1 < 0 ? 0 : (i0 + 1 > Min - 1 ? Min - 1 : i0 + 1);
    const float* xb = x + (size_t)b * Min * C;
    y[t] = xb[(size_t)ia * C + c] * (1.0f - w) + xb[(size_t)ic * C + c] * w;
}

__global__ void add_kernel(const float* __restrict__ a, const float* __restrict__ b,
                           float* __restrict__ c, size_t n)
{
    size_t t = (size_t)blockIdx.x * blockDim.x + threadIdx.x;
    if (t < n) c[t] = a[t] + b[t];
}

// ---------------------------------------------------------------------------
// Host orchestration
// ---------------------------------------------------------------------------
static inline unsigned cdiv(size_t a, size_t b) { return (unsigned)((a + b - 1) / b); }

struct Arena {
    char* base; size_t cap; size_t off;
    void* get(size_t bytes) {
        size_t a = (off + 255) & ~(size_t)255;
        off = a + bytes;
        return base + a;
    }
    float*    f(size_t n) { return (float*)get(n * sizeof(float)); }
    int*      i(size_t n) { return (int*)get(n * sizeof(int)); }
    _Float16* h(size_t n) { return (_Float16*)get(n * sizeof(_Float16)); }
};

extern "C" void kernel_launch(void* const* d_in, const int* in_sizes, int n_in,
                              void* d_out, int out_size, void* d_ws, size_t ws_size,
                              hipStream_t stream)
{
    (void)in_sizes; (void)n_in; (void)out_size;
    const int Bp = 2, N0 = 4096;

    const float* x = (const float*)d_in[0];
    const float* p = (const float*)d_in[1];
    // 270 parameter leaves from index 2, jax tree_flatten order (alpha keys):
    // lin_cls{b,w}=0,1  lin_in{b,w}=2,3  lin_mid{b,w}=4,5
    // pt_in=6  ptd1=30 ptd2=54 ptd3=78 ptd4=102  ptu1=126 ptu2=150 ptu3=174 ptu4=198
    // td1=222 td2=228 td3=234 td4=240  tu1=246 tu2=252 tu3=258 tu4=264
    // _ptb(24): lin1{l1b,l1w,l2b,l2w}=0..3 lin2=4..7 pt.k=8..11 pt.pe=12..15 pt.q=16..19 pt.v=20..23
    // _tr(6): beta=0 gamma=1 l1b=2 l1w=3 l2b=4 l2w=5
    auto L = [&](int i) -> const float* { return (const float*)d_in[2 + i]; };

    Arena A{ (char*)d_ws, ws_size, 0 };

    // ---- persistent buffers ----
    float* XIN0 = A.f((size_t)Bp * 4096 * 32);
    float* XIN  = A.f((size_t)Bp * 4096 * 32);
    float* XD1_0 = A.f((size_t)Bp * 1024 * 64);  float* XD1 = A.f((size_t)Bp * 1024 * 64);
    float* PD1   = A.f((size_t)Bp * 1024 * 3);
    float* XD2_0 = A.f((size_t)Bp * 256 * 128);  float* XD2 = A.f((size_t)Bp * 256 * 128);
    float* PD2   = A.f((size_t)Bp * 256 * 3);
    float* XD3_0 = A.f((size_t)Bp * 64 * 256);   float* XD3 = A.f((size_t)Bp * 64 * 256);
    float* PD3   = A.f((size_t)Bp * 64 * 3);
    float* XD4_0 = A.f((size_t)Bp * 16 * 512);   float* XD4 = A.f((size_t)Bp * 16 * 512);
    float* PD4   = A.f((size_t)Bp * 16 * 3);
    float* XMID  = A.f((size_t)Bp * 16 * 512);
    float* XU4_0 = A.f((size_t)Bp * 64 * 256);   float* XU4in = A.f((size_t)Bp * 64 * 256);
    float* XU4   = A.f((size_t)Bp * 64 * 256);   float* PU4   = A.f((size_t)Bp * 64 * 3);
    float* XU3_0 = A.f((size_t)Bp * 256 * 128);  float* XU3in = A.f((size_t)Bp * 256 * 128);
    float* XU3   = A.f((size_t)Bp * 256 * 128);  float* PU3   = A.f((size_t)Bp * 256 * 3);
    float* XU2_0 = A.f((size_t)Bp * 1024 * 64);  float* XU2in = A.f((size_t)Bp * 1024 * 64);
    float* XU2   = A.f((size_t)Bp * 1024 * 64);  float* PU2   = A.f((size_t)Bp * 1024 * 3);
    float* XU1_0 = A.f((size_t)Bp * 4096 * 32);  float* XU1in = A.f((size_t)Bp * 4096 * 32);
    float* XU1   = A.f((size_t)Bp * 4096 * 32);  float* PU1   = A.f((size_t)Bp * 4096 * 3);

    // Reusable GEMM pack region (stream serializes pack->gemm->next pack).
    // Max Xp: M=131072 rows (pe GEMMs), Kpad=32 -> 4M halfs. Max Wt: 512x512.
    _Float16* PACKX = A.h((size_t)4 * 1024 * 1024 + 4096);
    _Float16* PACKW = A.h((size_t)512 * 512 + 4096);

    const size_t mark = A.off;   // stage scratch starts here (reused per stage)

    auto gemm = [&](const float* X_, const float* W_, const float* bias,
                    const float* res, float* Y_, int M, int K, int Nd, int relu) {
        const int Kpad = ((K + 31) / 32) * 32;
        const int Mpad = ((M + 31) / 32) * 32;
        const int Npad = ((Nd + 15) / 16) * 16;
        size_t na = (size_t)Mpad * Kpad;
        size_t nb = (size_t)Npad * Kpad;
        pack_a_kernel<<<cdiv(na, 256), 256, 0, stream>>>(X_, PACKX, M, K, Kpad, na);
        pack_bt_kernel<<<cdiv(nb, 256), 256, 0, stream>>>(W_, PACKW, K, Nd, Kpad, nb);
        dim3 g(cdiv(M, 128), (unsigned)(Npad / 16));
        gemm_wmma_kernel<<<g, dim3(128), 0, stream>>>(PACKX, PACKW, bias, res, Y_,
                                                      M, Kpad, Nd, relu);
    };
    // proj: Linear(K->Cout) -> ReLU -> Linear(Cout->Cout) [+res]
    auto proj = [&](const float* X_, int M, int K, int Cout,
                    const float* w1, const float* b1, const float* w2, const float* b2,
                    const float* res, float* tmp, float* Y_) {
        gemm(X_, w1, b1, nullptr, tmp, M, K, Cout, 1);
        gemm(tmp, w2, b2, res, Y_, M, Cout, Cout, 0);
    };

    auto pt_block = [&](int pb, const float* xin, const float* pos,
                        int Np, int C, float* y) {
        A.off = mark;
        const int Mp = Bp * Np;
        int*   idx = A.i((size_t)Mp * KNN_K);
        float* x1  = A.f((size_t)Mp * C);
        float* tmp = A.f((size_t)Mp * C);
        float* qb  = A.f((size_t)Mp * C);
        float* kb  = A.f((size_t)Mp * C);
        float* vb  = A.f((size_t)Mp * C);
        float* rel = A.f((size_t)Mp * KNN_K * 3);
        float* peT = A.f((size_t)Mp * KNN_K * C);
        float* pe  = A.f((size_t)Mp * KNN_K * C);
        float* o   = A.f((size_t)Mp * C);

        knn_kernel<<<cdiv(Mp, 128), 128, 0, stream>>>(xin, Bp, Np, C, Np, 1, idx);
        proj(xin, Mp, C, C, L(pb + 1),  L(pb + 0),  L(pb + 3),  L(pb + 2),  nullptr, tmp, x1);  // lin1
        proj(x1,  Mp, C, C, L(pb + 17), L(pb + 16), L(pb + 19), L(pb + 18), nullptr, tmp, qb);  // q
        proj(x1,  Mp, C, C, L(pb + 9),  L(pb + 8),  L(pb + 11), L(pb + 10), nullptr, tmp, kb);  // k
        proj(x1,  Mp, C, C, L(pb + 21), L(pb + 20), L(pb + 23), L(pb + 22), nullptr, tmp, vb);  // v
        relpos_kernel<<<cdiv((size_t)Mp * KNN_K * 3, 256), 256, 0, stream>>>(pos, idx, rel, Bp, Np);
        proj(rel, Mp * KNN_K, 3, C, L(pb + 13), L(pb + 12), L(pb + 15), L(pb + 14), nullptr, peT, pe); // pe
        attn_kernel<<<cdiv((size_t)Mp * C, 256), 256, 0, stream>>>(qb, kb, vb, pe, idx, o, Bp, Np, C);
        proj(o, Mp, C, C, L(pb + 5), L(pb + 4), L(pb + 7), L(pb + 6), xin, tmp, y);             // lin2 + res
    };

    auto tdown = [&](int tb, const float* xin, const float* pos,
                     int Nn, int Cin, int Cout, int n2, float* xout, float* pout) {
        A.off = mark;
        const int stride = Nn / n2;
        const int Mq = Bp * n2, Mr = Mq * KNN_K;
        int*   idx = A.i((size_t)Mq * KNN_K);
        float* X2  = A.f((size_t)Mr * Cin);
        float* H   = A.f((size_t)Mr * Cout);
        float* Hr  = A.f((size_t)Mr * Cout);
        float* H2  = A.f((size_t)Mr * Cout);
        float* mu  = A.f(Cout);
        float* rs  = A.f(Cout);
        knn_kernel<<<cdiv(Mq, 128), 128, 0, stream>>>(xin, Bp, Nn, Cin, n2, stride, idx);
        gather_kernel<<<cdiv((size_t)Mr * Cin, 256), 256, 0, stream>>>(xin, idx, X2, Bp, Nn, n2, Cin);
        gemm(X2, L(tb + 3), L(tb + 2), nullptr, H, Mr, Cin, Cout, 0);
        colstats_kernel<<<Cout, 256, 0, stream>>>(H, Mr, Cout, mu, rs);
        bnrelu_kernel<<<cdiv((size_t)Mr * Cout, 256), 256, 0, stream>>>(H, mu, rs, L(tb + 1), L(tb + 0), Hr, Mr, Cout);
        gemm(Hr, L(tb + 5), L(tb + 4), nullptr, H2, Mr, Cout, Cout, 0);
        maxk_kernel<<<cdiv((size_t)Mq * Cout, 256), 256, 0, stream>>>(H2, xout, Mq, Cout);
        possub_kernel<<<cdiv((size_t)Bp * n2 * 3, 256), 256, 0, stream>>>(pos, pout, Bp, Nn, n2, stride);
    };

    auto tup = [&](int tb, const float* xin, const float* pos,
                   int Min, int Cin, int Cout, float* xup, float* pup) {
        A.off = mark;
        const int Mr = Bp * Min;
        float* H  = A.f((size_t)Mr * Cout);
        float* Hr = A.f((size_t)Mr * Cout);
        float* H2 = A.f((size_t)Mr * Cout);
        float* mu = A.f(Cout);
        float* rs = A.f(Cout);
        gemm(xin, L(tb + 3), L(tb + 2), nullptr, H, Mr, Cin, Cout, 0);
        colstats_kernel<<<Cout, 256, 0, stream>>>(H, Mr, Cout, mu, rs);
        bnrelu_kernel<<<cdiv((size_t)Mr * Cout, 256), 256, 0, stream>>>(H, mu, rs, L(tb + 1), L(tb + 0), Hr, Mr, Cout);
        gemm(Hr, L(tb + 5), L(tb + 4), nullptr, H2, Mr, Cout, Cout, 0);
        upsample4_kernel<<<cdiv((size_t)Bp * Min * 4 * Cout, 256), 256, 0, stream>>>(H2, xup, Bp, Min, Cout);
        upsample4_kernel<<<cdiv((size_t)Bp * Min * 4 * 3, 256), 256, 0, stream>>>(pos, pup, Bp, Min, 3);
    };

    auto addv = [&](const float* a, const float* b, float* c, size_t n) {
        add_kernel<<<cdiv(n, 256), 256, 0, stream>>>(a, b, c, n);
    };

    // ---- forward pass ----
    gemm(x, L(3), L(2), nullptr, XIN0, Bp * N0, 6, 32, 0);          // lin_in
    pt_block(6,   XIN0, p,   N0,   32, XIN);                        // pt_in
    tdown(222, XIN, p,    4096, 32,  64, 1024, XD1_0, PD1);         // td1
    pt_block(30,  XD1_0, PD1, 1024, 64, XD1);                       // ptd1
    tdown(228, XD1, PD1, 1024, 64, 128, 256, XD2_0, PD2);           // td2
    pt_block(54,  XD2_0, PD2, 256, 128, XD2);                       // ptd2
    tdown(234, XD2, PD2, 256, 128, 256, 64, XD3_0, PD3);            // td3
    pt_block(78,  XD3_0, PD3, 64,  256, XD3);                       // ptd3
    tdown(240, XD3, PD3, 64, 256, 512, 16, XD4_0, PD4);             // td4
    pt_block(102, XD4_0, PD4, 16,  512, XD4);                       // ptd4
    gemm(XD4, L(5), L(4), nullptr, XMID, Bp * 16, 512, 512, 0);     // lin_mid
    tup(264, XMID, PD4, 16, 512, 256, XU4_0, PU4);                  // tu4
    addv(XU4_0, XD3, XU4in, (size_t)Bp * 64 * 256);
    pt_block(198, XU4in, PU4, 64,  256, XU4);                       // ptu4
    tup(258, XU4, PU4, 64, 256, 128, XU3_0, PU3);                   // tu3
    addv(XU3_0, XD2, XU3in, (size_t)Bp * 256 * 128);
    pt_block(174, XU3in, PU3, 256, 128, XU3);                       // ptu3
    tup(252, XU3, PU3, 256, 128, 64, XU2_0, PU2);                   // tu2
    addv(XU2_0, XD1, XU2in, (size_t)Bp * 1024 * 64);
    pt_block(150, XU2in, PU2, 1024, 64, XU2);                       // ptu2
    tup(246, XU2, PU2, 1024, 64, 32, XU1_0, PU1);                   // tu1
    addv(XU1_0, XIN, XU1in, (size_t)Bp * 4096 * 32);
    pt_block(126, XU1in, PU1, 4096, 32, XU1);                       // ptu1
    gemm(XU1, L(1), L(0), nullptr, (float*)d_out, Bp * 4096, 32, 13, 0); // lin_cls
}
